// NEXUSLite_24532853194971
// MI455X (gfx1250) — matrix operand order repaired and checked by reading, not compiled, not run on hardware
//
#include <hip/hip_runtime.h>

// ---------------------------------------------------------------------------
// DaViT block (SpatialBlock + ChannelBlock) for gfx1250 (MI455X).
// Dense matmuls -> v_wmma_f32_16x16x32_f16 (f16 in, f32 acc).
// GEMM weight tiles staged into LDS via the Tensor Data Mover
// (tensor_load_to_lds + s_wait_tensorcnt).
// B=8, H=W=128, C=192, HEADS=6, head_dim=32, window=8 (ws2=64), HIDDEN=384.
// ---------------------------------------------------------------------------

typedef __attribute__((ext_vector_type(16))) _Float16 v16h;
typedef __attribute__((ext_vector_type(8)))  _Float16 half8;
typedef __attribute__((ext_vector_type(8)))  float    v8f;
typedef __attribute__((ext_vector_type(4)))  unsigned int u32x4;
typedef __attribute__((ext_vector_type(8)))  int      i32x8;
typedef __attribute__((ext_vector_type(4)))  int      i32x4;

#define M_TOKENS 131072ULL   // 8 * 128 * 128
#define C_DIM    192
#define HID_DIM  384
#define QKV_DIM  576

// ---------------------------------------------------------------------------
// f32 -> f16 conversion (weight packing)
// ---------------------------------------------------------------------------
__global__ void k_cvt_f16(const float* __restrict__ in, _Float16* __restrict__ out, int n) {
    int i = blockIdx.x * blockDim.x + threadIdx.x;
    if (i < n) out[i] = (_Float16)in[i];
}

__global__ void k_zero_f32(float* __restrict__ p, int n) {
    int i = blockIdx.x * blockDim.x + threadIdx.x;
    if (i < n) p[i] = 0.0f;
}

// ---------------------------------------------------------------------------
// LayerNorm over C=192, one wave (32 lanes) per row, f16 output.
// ---------------------------------------------------------------------------
__global__ void k_layernorm(const float* __restrict__ x, const float* __restrict__ w,
                            const float* __restrict__ b, _Float16* __restrict__ y) {
    int wave = threadIdx.x >> 5;
    int lane = threadIdx.x & 31;
    size_t row = (size_t)blockIdx.x * 8 + wave;
    const float* xr = x + row * C_DIM;
    float v[6], s = 0.0f, sq = 0.0f;
#pragma unroll
    for (int i = 0; i < 6; ++i) {
        v[i] = xr[lane + i * 32];
        s += v[i]; sq += v[i] * v[i];
    }
#pragma unroll
    for (int m = 16; m > 0; m >>= 1) {
        s  += __shfl_xor(s,  m);
        sq += __shfl_xor(sq, m);
    }
    float mean = s * (1.0f / C_DIM);
    float var  = sq * (1.0f / C_DIM) - mean * mean;
    float inv  = rsqrtf(var + 1e-5f);
    _Float16* yr = y + row * C_DIM;
#pragma unroll
    for (int i = 0; i < 6; ++i) {
        int c = lane + i * 32;
        yr[c] = (_Float16)((v[i] - mean) * inv * w[c] + b[c]);
    }
}

// ---------------------------------------------------------------------------
// WMMA GEMM: out[M,N] = A[M,K] (f16, row-major) @ W[N,K]^T (f16) + bias.
// W stored row-major [N,K] == B matrix [K,N] column-major, matching CDNA5
// B-fragment striping (contiguous 16B loads per lane).
// Per block: TDM loads the 32-column weight slab (32 rows x K halves) into
// LDS once; 8 waves each own one 16x16 D tile (block tile 64x32).
// EPI 0: store f16(acc+bias).    EPI 1: store f32 resid + ls[col]*(acc+bias).
// ---------------------------------------------------------------------------
template<int EPI>
__global__ void k_gemm(const _Float16* __restrict__ A, const _Float16* __restrict__ W,
                       const float* __restrict__ bias, const float* __restrict__ resid,
                       const float* __restrict__ ls, void* __restrict__ outp,
                       int K, int N) {
    __shared__ __align__(16) _Float16 wlds[32 * HID_DIM];   // max 32 x 384 f16 = 24KB

    // ---- Tensor Data Mover: stage W[blockIdx.x*32 .. +31][0..K) into LDS.
    // D# (ISA ch.8): 2D tensor of dwords, tile = (K/2) x 32, stride = K/2.
    {
        const unsigned K2 = (unsigned)(K >> 1);              // dwords per row
        const unsigned Nr = (unsigned)N;                     // total rows
        unsigned long long gaddr =
            (unsigned long long)(uintptr_t)(W + (size_t)blockIdx.x * 32 * (size_t)K);
        unsigned laddr = (unsigned)(uintptr_t)(&wlds[0]);    // LDS offset = low 32 bits
        if (threadIdx.x < 32) {
            u32x4 g0;
            g0[0] = 1u;                                       // count = 1 descriptor
            g0[1] = laddr;                                    // lds_addr
            g0[2] = (unsigned)gaddr;                          // global_addr[31:0]
            g0[3] = (unsigned)(gaddr >> 32) | (2u << 30);     // addr[56:32] | type=2
            i32x8 g1;
            g1[0] = (int)(2u << 16);                          // data_size = 4 bytes
            g1[1] = (int)((K2 & 0xFFFFu) << 16);              // tensor_dim0 lo
            g1[2] = (int)((K2 >> 16) | ((Nr & 0xFFFFu) << 16)); // dim0 hi | dim1 lo
            g1[3] = (int)((Nr >> 16) | (K2 << 16));           // dim1 hi | tile_dim0
            g1[4] = 32;                                       // tile_dim1 = 32 rows
            g1[5] = (int)K2;                                  // tensor_dim0_stride lo
            g1[6] = 0;
            g1[7] = 0;
            i32x4 gz  = {0, 0, 0, 0};
            i32x8 gz8 = {0, 0, 0, 0, 0, 0, 0, 0};
            __builtin_amdgcn_tensor_load_to_lds(g0, g1, gz, gz, gz8, 0);
            __builtin_amdgcn_s_wait_tensorcnt(0);
        }
    }
    __syncthreads();

    int wave = threadIdx.x >> 5;
    int lane = threadIdx.x & 31;
    int ln15 = lane & 15;
    int hi   = lane >> 4;
    int tm = wave >> 1, tn = wave & 1;

    size_t rowA = (size_t)blockIdx.y * 64 + tm * 16 + ln15;
    int    col  = blockIdx.x * 32 + tn * 16 + ln15;

    const _Float16* Ap = A + rowA * (size_t)K;
    const _Float16* Wp = wlds + (size_t)(tn * 16 + ln15) * (size_t)K;

    v8f acc = {};
    for (int k0 = 0; k0 < K; k0 += 32) {
        // A fragment: lane-half hi covers K = k0+hi*8..+7 and k0+16+hi*8..+7
        half8 alo = *(const half8*)(Ap + k0 + hi * 8);
        half8 ahi = *(const half8*)(Ap + k0 + 16 + hi * 8);
        // B fragment (from LDS): lane-half hi covers K = k0+hi*8..+7 and k0+16+hi*8..+7
        half8 blo = *(const half8*)(Wp + k0 + hi * 8);
        half8 bhi = *(const half8*)(Wp + k0 + 16 + hi * 8);
        v16h af, bf;
#pragma unroll
        for (int i = 0; i < 8; ++i) {
            af[i] = alo[i]; af[i + 8] = ahi[i];
            bf[i] = blo[i]; bf[i + 8] = bhi[i];
        }
        acc = __builtin_amdgcn_wmma_f32_16x16x32_f16(
            false, af, false, bf, (short)0, acc, false, false);
    }

    float bv = bias[col];
    size_t rbase = (size_t)blockIdx.y * 64 + tm * 16 + hi * 8;
#pragma unroll
    for (int r = 0; r < 8; ++r) {
        size_t idx = (rbase + r) * (size_t)N + col;
        float v = acc[r] + bv;
        if constexpr (EPI == 0) {
            ((_Float16*)outp)[idx] = (_Float16)v;
        } else {
            ((float*)outp)[idx] = resid[idx] + ls[col] * v;
        }
    }
}

// ---------------------------------------------------------------------------
// Window attention: one 256-thread block per (window, head).
// qkv: [M, 576] f16 (q|k|v each 192 = 6 heads * 32). out: [M, 192] f16.
// S = q k^T * scale + alibi -> softmax -> O = P v, all via WMMA f16.
// ---------------------------------------------------------------------------
__global__ void k_attn(const _Float16* __restrict__ qkv, _Float16* __restrict__ out) {
    __shared__ __align__(16) _Float16 qs[64 * 32];
    __shared__ __align__(16) _Float16 ks[64 * 32];
    __shared__ __align__(16) _Float16 vts[32 * 64];   // v transposed: [d][p]
    __shared__ __align__(16) float    Ss[64 * 64];
    __shared__ __align__(16) _Float16 Ps[64 * 64];

    int wh   = blockIdx.x;
    int head = wh % 6;
    int win  = wh / 6;
    int b    = win >> 8;          // 256 windows per batch image
    int wrem = win & 255;
    int wy   = wrem >> 4;
    int wx   = wrem & 15;

    int t = threadIdx.x;
    // ---- load q,k,v tiles: thread t -> position p = t/4, 8 dims d0..d0+7
    {
        int p  = t >> 2;
        int d0 = (t & 3) * 8;
        int i = p >> 3, j = p & 7;
        size_t tok = (size_t)b * 16384 + (size_t)((wy * 8 + i) * 128 + (wx * 8 + j));
        const _Float16* base = qkv + tok * QKV_DIM + head * 32;
        half8 qv = *(const half8*)(base + d0);
        half8 kv = *(const half8*)(base + 192 + d0);
        half8 vv = *(const half8*)(base + 384 + d0);
        *(half8*)(qs + p * 32 + d0) = qv;
        *(half8*)(ks + p * 32 + d0) = kv;
#pragma unroll
        for (int dd = 0; dd < 8; ++dd) vts[(d0 + dd) * 64 + p] = vv[dd];
    }
    __syncthreads();

    int wave = t >> 5, lane = t & 31;
    int ln15 = lane & 15, hi = lane >> 4;
    float slope = -exp2f(-(4.0f / 3.0f) * (float)(head + 1));
    const float scale = 0.17677669529663687f; // 32^-0.5

    // ---- S = q @ k^T : 16 tiles of 16x16, 2 per wave, K = 32 (one WMMA)
    for (int tt = wave; tt < 16; tt += 8) {
        int ti = tt >> 2, tj = tt & 3;
        const _Float16* ap = qs + (ti * 16 + ln15) * 32;
        const _Float16* bp = ks + (tj * 16 + ln15) * 32;
        half8 alo = *(const half8*)(ap + hi * 8);
        half8 ahi = *(const half8*)(ap + 16 + hi * 8);
        half8 blo = *(const half8*)(bp + hi * 8);
        half8 bhi = *(const half8*)(bp + 16 + hi * 8);
        v16h af, bf;
#pragma unroll
        for (int i = 0; i < 8; ++i) {
            af[i] = alo[i]; af[i + 8] = ahi[i];
            bf[i] = blo[i]; bf[i + 8] = bhi[i];
        }
        v8f acc = {};
        acc = __builtin_amdgcn_wmma_f32_16x16x32_f16(
            false, af, false, bf, (short)0, acc, false, false);
#pragma unroll
        for (int r = 0; r < 8; ++r) {
            int ii = ti * 16 + hi * 8 + r;
            int jj = tj * 16 + ln15;
            Ss[ii * 64 + jj] = acc[r] * scale + slope * fabsf((float)(ii - jj));
        }
    }
    __syncthreads();

    // ---- softmax over rows (64 rows, one thread each)
    if (t < 64) {
        float mx = -1e30f;
        for (int j = 0; j < 64; ++j) mx = fmaxf(mx, Ss[t * 64 + j]);
        float sum = 0.0f;
        for (int j = 0; j < 64; ++j) sum += __expf(Ss[t * 64 + j] - mx);
        float inv = 1.0f / sum;
        for (int j = 0; j < 64; ++j)
            Ps[t * 64 + j] = (_Float16)(__expf(Ss[t * 64 + j] - mx) * inv);
    }
    __syncthreads();

    // ---- O = P @ V : 64x32 out = 8 tiles, one per wave, K = 64 (2 WMMAs)
    {
        int ti = wave >> 1, tn = wave & 1;
        v8f acc = {};
        for (int kb = 0; kb < 64; kb += 32) {
            const _Float16* ap = Ps + (ti * 16 + ln15) * 64 + kb;
            const _Float16* bp = vts + (tn * 16 + ln15) * 64 + kb;
            half8 alo = *(const half8*)(ap + hi * 8);
            half8 ahi = *(const half8*)(ap + 16 + hi * 8);
            half8 blo = *(const half8*)(bp + hi * 8);
            half8 bhi = *(const half8*)(bp + 16 + hi * 8);
            v16h af, bf;
#pragma unroll
            for (int i = 0; i < 8; ++i) {
                af[i] = alo[i]; af[i + 8] = ahi[i];
                bf[i] = blo[i]; bf[i + 8] = bhi[i];
            }
            acc = __builtin_amdgcn_wmma_f32_16x16x32_f16(
                false, af, false, bf, (short)0, acc, false, false);
        }
#pragma unroll
        for (int r = 0; r < 8; ++r) {
            int p = ti * 16 + hi * 8 + r;
            int i = p >> 3, j = p & 7;
            size_t tok = (size_t)b * 16384 + (size_t)((wy * 8 + i) * 128 + (wx * 8 + j));
            out[tok * C_DIM + head * 32 + tn * 16 + ln15] = (_Float16)acc[r];
        }
    }
}

// ---------------------------------------------------------------------------
// Depthwise 3x3 conv (pad 1) + bias + exact GELU.  h,g: [B,128,128,384] f16.
// ---------------------------------------------------------------------------
__global__ void k_dwconv_gelu(const _Float16* __restrict__ h, const float* __restrict__ w,
                              const float* __restrict__ bias, _Float16* __restrict__ g) {
    size_t idx = (size_t)blockIdx.x * blockDim.x + threadIdx.x;
    if (idx >= (size_t)8 * 16384 * HID_DIM) return;
    int c = (int)(idx % HID_DIM);
    size_t r = idx / HID_DIM;
    int xx = (int)(r % 128);
    size_t r2 = r / 128;
    int yy = (int)(r2 % 128);
    int bb = (int)(r2 / 128);
    float acc = bias[c];
#pragma unroll
    for (int ky = 0; ky < 3; ++ky) {
#pragma unroll
        for (int kx = 0; kx < 3; ++kx) {
            int y2 = yy + ky - 1, x2 = xx + kx - 1;
            if (y2 >= 0 && y2 < 128 && x2 >= 0 && x2 < 128) {
                size_t src = (((size_t)bb * 128 + y2) * 128 + x2) * HID_DIM + c;
                acc += (float)h[src] * w[c * 9 + ky * 3 + kx];
            }
        }
    }
    float ge = 0.5f * acc * (1.0f + erff(acc * 0.70710678118654752f));
    g[idx] = (_Float16)ge;
}

// ---------------------------------------------------------------------------
// ECA: partial sums over tokens -> gate -> fused residual apply.
// ---------------------------------------------------------------------------
__global__ void k_gap_partial(const _Float16* __restrict__ ln, float* __restrict__ mean) {
    int b = blockIdx.x, chunk = blockIdx.y, c = threadIdx.x;   // block 192 threads
    float s = 0.0f;
    size_t base = ((size_t)b * 16384 + (size_t)chunk * 512) * C_DIM + c;
    for (int n = 0; n < 512; ++n) s += (float)ln[base + (size_t)n * C_DIM];
    atomicAdd(&mean[b * C_DIM + c], s);
}

__global__ void k_eca_gate(const float* __restrict__ mean, const float* __restrict__ ew,
                           float* __restrict__ gate) {
    int t = blockIdx.x * blockDim.x + threadIdx.x;
    if (t >= 8 * C_DIM) return;
    int b = t / C_DIM, c = t % C_DIM;
    float ym1 = (c > 0)         ? mean[b * C_DIM + c - 1] * (1.0f / 16384.0f) : 0.0f;
    float y0  =                   mean[b * C_DIM + c]     * (1.0f / 16384.0f);
    float yp1 = (c < C_DIM - 1) ? mean[b * C_DIM + c + 1] * (1.0f / 16384.0f) : 0.0f;
    float yc = ew[0] * ym1 + ew[1] * y0 + ew[2] * yp1;
    gate[t] = 1.0f / (1.0f + __expf(-yc));
}

__global__ void k_eca_apply(float* __restrict__ x, const _Float16* __restrict__ ln,
                            const float* __restrict__ gate, const float* __restrict__ cls1) {
    size_t idx = (size_t)blockIdx.x * blockDim.x + threadIdx.x;
    if (idx >= M_TOKENS * C_DIM) return;
    int c = (int)(idx % C_DIM);
    size_t row = idx / C_DIM;
    int b = (int)(row >> 14);   // 16384 tokens per batch
    x[idx] = x[idx] + cls1[c] * (float)ln[idx] * gate[b * C_DIM + c];
}

// ---------------------------------------------------------------------------
// Host orchestration
// ---------------------------------------------------------------------------
extern "C" void kernel_launch(void* const* d_in, const int* in_sizes, int n_in,
                              void* d_out, int out_size, void* d_ws, size_t ws_size,
                              hipStream_t stream) {
    const float* x      = (const float*)d_in[0];
    const float* ln1_w  = (const float*)d_in[1];
    const float* ln1_b  = (const float*)d_in[2];
    const float* qkv_w  = (const float*)d_in[3];
    const float* qkv_b  = (const float*)d_in[4];
    const float* proj_w = (const float*)d_in[5];
    const float* proj_b = (const float*)d_in[6];
    const float* ls1    = (const float*)d_in[7];
    const float* ls2    = (const float*)d_in[8];
    const float* ln2_w  = (const float*)d_in[9];
    const float* ln2_b  = (const float*)d_in[10];
    const float* fc1_w  = (const float*)d_in[11];
    const float* fc1_b  = (const float*)d_in[12];
    const float* dw_w   = (const float*)d_in[13];
    const float* dw_b   = (const float*)d_in[14];
    const float* fc2_w  = (const float*)d_in[15];
    const float* fc2_b  = (const float*)d_in[16];
    const float* cln1_w = (const float*)d_in[17];
    const float* cln1_b = (const float*)d_in[18];
    const float* eca_w  = (const float*)d_in[19];
    const float* cls1   = (const float*)d_in[20];
    const float* cls2   = (const float*)d_in[21];
    const float* cln2_w = (const float*)d_in[22];
    const float* cln2_b = (const float*)d_in[23];
    const float* cfc1_w = (const float*)d_in[24];
    const float* cfc1_b = (const float*)d_in[25];
    const float* cdw_w  = (const float*)d_in[26];
    const float* cdw_b  = (const float*)d_in[27];
    const float* cfc2_w = (const float*)d_in[28];
    const float* cfc2_b = (const float*)d_in[29];
    float* out = (float*)d_out;

    char* ws = (char*)d_ws;
    size_t off = 0;
    auto alloc = [&](size_t bytes) {
        size_t r = off;
        off += (bytes + 255) & ~(size_t)255;
        return r;
    };
    const size_t M = M_TOKENS;
    _Float16* whqkv  = (_Float16*)(ws + alloc((size_t)QKV_DIM * C_DIM * 2));
    _Float16* whproj = (_Float16*)(ws + alloc((size_t)C_DIM * C_DIM * 2));
    _Float16* whfc1  = (_Float16*)(ws + alloc((size_t)HID_DIM * C_DIM * 2));
    _Float16* whfc2  = (_Float16*)(ws + alloc((size_t)C_DIM * HID_DIM * 2));
    _Float16* whcfc1 = (_Float16*)(ws + alloc((size_t)HID_DIM * C_DIM * 2));
    _Float16* whcfc2 = (_Float16*)(ws + alloc((size_t)C_DIM * HID_DIM * 2));
    float*    meanb  = (float*)(ws + alloc(8 * C_DIM * 4));
    float*    gateb  = (float*)(ws + alloc(8 * C_DIM * 4));
    _Float16* t_ln   = (_Float16*)(ws + alloc(M * C_DIM * 2));
    _Float16* qkvbuf = (_Float16*)(ws + alloc(M * QKV_DIM * 2));   // reused for FFN hidden h
    _Float16* attnb  = (_Float16*)(ws + alloc(M * C_DIM * 2));
    float*    x_acc  = (float*)(ws + alloc(M * C_DIM * 4));
    _Float16* gbuf   = (_Float16*)(ws + alloc(M * HID_DIM * 2));
    _Float16* hbuf   = qkvbuf;

    // ---- pack weights to f16 (W stays row-major [N,K] = WMMA B layout)
    auto cvt = [&](const float* src, _Float16* dst, int n) {
        k_cvt_f16<<<(n + 255) / 256, 256, 0, stream>>>(src, dst, n);
    };
    cvt(qkv_w,  whqkv,  QKV_DIM * C_DIM);
    cvt(proj_w, whproj, C_DIM * C_DIM);
    cvt(fc1_w,  whfc1,  HID_DIM * C_DIM);
    cvt(fc2_w,  whfc2,  C_DIM * HID_DIM);
    cvt(cfc1_w, whcfc1, HID_DIM * C_DIM);
    cvt(cfc2_w, whcfc2, C_DIM * HID_DIM);

    dim3 blk(256);
    dim3 gLN((unsigned)(M / 8));
    dim3 gGemmC((unsigned)(C_DIM / 32), (unsigned)(M / 64));
    dim3 gGemmQKV((unsigned)(QKV_DIM / 32), (unsigned)(M / 64));
    dim3 gGemmH((unsigned)(HID_DIM / 32), (unsigned)(M / 64));

    // ======================= SpatialBlock =======================
    // ln1 -> qkv -> window attn -> proj (+resid, ls1)
    k_layernorm<<<gLN, blk, 0, stream>>>(x, ln1_w, ln1_b, t_ln);
    k_gemm<0><<<gGemmQKV, blk, 0, stream>>>(t_ln, whqkv, qkv_b, nullptr, nullptr,
                                            qkvbuf, C_DIM, QKV_DIM);
    k_attn<<<dim3(2048 * 6), blk, 0, stream>>>(qkvbuf, attnb);
    k_gemm<1><<<gGemmC, blk, 0, stream>>>(attnb, whproj, proj_b, x, ls1,
                                          x_acc, C_DIM, C_DIM);
    // ln2 -> fc1 -> dwconv+gelu -> fc2 (+resid, ls2)
    k_layernorm<<<gLN, blk, 0, stream>>>(x_acc, ln2_w, ln2_b, t_ln);
    k_gemm<0><<<gGemmH, blk, 0, stream>>>(t_ln, whfc1, fc1_b, nullptr, nullptr,
                                          hbuf, C_DIM, HID_DIM);
    k_dwconv_gelu<<<dim3((unsigned)(M * HID_DIM / 256)), blk, 0, stream>>>(hbuf, dw_w, dw_b, gbuf);
    k_gemm<1><<<gGemmC, blk, 0, stream>>>(gbuf, whfc2, fc2_b, x_acc, ls2,
                                          x_acc, HID_DIM, C_DIM);

    // ======================= ChannelBlock =======================
    // cln1 -> ECA gate -> fused residual (cls1)
    k_layernorm<<<gLN, blk, 0, stream>>>(x_acc, cln1_w, cln1_b, t_ln);
    k_zero_f32<<<(8 * C_DIM + 255) / 256, 256, 0, stream>>>(meanb, 8 * C_DIM);
    k_gap_partial<<<dim3(8, 32), dim3(C_DIM), 0, stream>>>(t_ln, meanb);
    k_eca_gate<<<(8 * C_DIM + 255) / 256, 256, 0, stream>>>(meanb, eca_w, gateb);
    k_eca_apply<<<dim3((unsigned)(M * C_DIM / 256)), blk, 0, stream>>>(x_acc, t_ln, gateb, cls1);
    // cln2 -> cfc1 -> dwconv+gelu -> cfc2 (+resid, cls2) -> d_out
    k_layernorm<<<gLN, blk, 0, stream>>>(x_acc, cln2_w, cln2_b, t_ln);
    k_gemm<0><<<gGemmH, blk, 0, stream>>>(t_ln, whcfc1, cfc1_b, nullptr, nullptr,
                                          hbuf, C_DIM, HID_DIM);
    k_dwconv_gelu<<<dim3((unsigned)(M * HID_DIM / 256)), blk, 0, stream>>>(hbuf, cdw_w, cdw_b, gbuf);
    k_gemm<1><<<gGemmC, blk, 0, stream>>>(gbuf, whcfc2, cfc2_b, x_acc, cls2,
                                          out, HID_DIM, C_DIM);
    (void)in_sizes; (void)n_in; (void)out_size; (void)ws_size;
}
